// ProbMultiHeadAttention_68478958568062
// MI455X (gfx1250) — matrix operand order repaired and checked by reading, not compiled, not run on hardware
//
#include <hip/hip_runtime.h>

// ---------------------------------------------------------------------------
// ProbMultiHeadAttention for MI455X (gfx1250), bf16 WMMA + f32 accumulate.
// B=1, L=2048, D=512, H=8, Dh=64.  Memory-bound (~430MB @ 23.3TB/s), so all
// matrix math is v_wmma_f32_16x16x32_bf16. No fast f32->bf16 instruction is
// available on this toolchain, so activations/weights are converted to bf16
// ONCE (at GEMM epilogues / a one-time weight transpose) and every staging
// loop afterwards is a pure b128 copy with zero conversion VALU.
// ---------------------------------------------------------------------------

#define L_ 2048
#define D_ 512
#define H_ 8
#define DH_ 64
#define SCALE_ 0.125f  // 1/sqrt(64)

typedef __attribute__((ext_vector_type(4)))  float          v4f;
typedef __attribute__((ext_vector_type(8)))  float          v8f;
typedef __attribute__((ext_vector_type(4)))  unsigned int   v4u;
typedef __attribute__((ext_vector_type(16))) __bf16         v16bf;
typedef __attribute__((ext_vector_type(8)))  unsigned short v8us;
typedef __attribute__((ext_vector_type(16))) unsigned short v16us;

__device__ __forceinline__ unsigned short f2bf(float f) {
  unsigned u = __builtin_bit_cast(unsigned, f);
  u += 0x7FFFu + ((u >> 16) & 1u);  // round-to-nearest-even
  return (unsigned short)(u >> 16);
}

__device__ __forceinline__ unsigned pk2(float a, float b) {
  return (unsigned)f2bf(a) | ((unsigned)f2bf(b) << 16);
}

// Convert 16 contiguous f32 -> 16 packed bf16 in LDS (two ds_store_b128).
__device__ __forceinline__ void cvt16(const float* __restrict__ g,
                                      unsigned short* l) {
  v4f a = *(const v4f*)(g);
  v4f b = *(const v4f*)(g + 4);
  v4f c = *(const v4f*)(g + 8);
  v4f d = *(const v4f*)(g + 12);
  v4u u0 = {pk2(a[0], a[1]), pk2(a[2], a[3]), pk2(b[0], b[1]), pk2(b[2], b[3])};
  v4u u1 = {pk2(c[0], c[1]), pk2(c[2], c[3]), pk2(d[0], d[1]), pk2(d[2], d[3])};
  *(v4u*)l = u0;
  *(v4u*)(l + 8) = u1;
}

// Copy 16 contiguous bf16 (32B) global -> LDS, zero VALU.
__device__ __forceinline__ void cpy16(const unsigned short* __restrict__ g,
                                      unsigned short* l) {
  v4u u0 = *(const v4u*)(g);
  v4u u1 = *(const v4u*)(g + 8);
  *(v4u*)l = u0;
  *(v4u*)(l + 8) = u1;
}

// A fragment: 16x32 bf16, lane r(0..15)/hi: elems 0..15 = K[hi*16..hi*16+15]
__device__ __forceinline__ v16bf ldA(const unsigned short* base, int stride,
                                     int m, int r, int hi, int ks) {
  const unsigned short* p = base + (size_t)(m + r) * stride + ks + hi * 16;
  v8us lo = *(const v8us*)p;
  v8us hh = *(const v8us*)(p + 8);
  v16us u = __builtin_shufflevector(lo, hh, 0,1,2,3,4,5,6,7,8,9,10,11,12,13,14,15);
  return __builtin_bit_cast(v16bf, u);
}

// B fragment: 32x16 bf16 from LDS stored [n][k]: lane c = column, elems = K[hi*16 + e]
__device__ __forceinline__ v16bf ldB(const unsigned short* base, int stride,
                                     int n, int c, int hi, int ks) {
  v16us u = *(const v16us*)(base + (size_t)(n + c) * stride + ks + hi * 16);
  return __builtin_bit_cast(v16bf, u);
}

// ---------------------------------------------------------------------------
// One-time weight transpose+convert: WT[n][k] = bf16(W[k][n]), 512x512.
// ---------------------------------------------------------------------------
__global__ __launch_bounds__(256) void transpose_cvt_kernel(
    const float* __restrict__ W, unsigned short* __restrict__ WT) {
  __shared__ unsigned short tile[32][33];
  const int bx = blockIdx.x * 32, by = blockIdx.y * 32;
  const int tx = threadIdx.x & 31, ty = (threadIdx.x >> 5) * 4;
  #pragma unroll
  for (int j = 0; j < 4; ++j)
    tile[ty + j][tx] = f2bf(W[(size_t)(by + ty + j) * D_ + bx + tx]);
  __syncthreads();
  #pragma unroll
  for (int j = 0; j < 4; ++j)
    WT[(size_t)(bx + ty + j) * D_ + by + tx] = tile[tx][ty + j];
}

// ---------------------------------------------------------------------------
// Tiled GEMM: C[M,N] = alpha * A[M,K] @ B[K,N] + bias
//   ABF: A is bf16 (ws) else f32;  BT: B stored transposed [N][K] bf16;
//   CBF: C stored bf16 (ws) else f32.
// Block 128x64, BK=32, 256 threads = 8 waves in 4x2, 32x32 tile per wave.
// ---------------------------------------------------------------------------
#define GBM 128
#define GBN 64
#define GBK 32

template <bool ABF, bool BT, bool CBF>
__global__ __launch_bounds__(256) void gemm_bf16_kernel(
    const void* __restrict__ Av, int lda,
    const unsigned short* __restrict__ B, int ldb,
    void* __restrict__ Cv, int ldc,
    const float* __restrict__ bias,
    float alpha, int M, int N, int K) {
  __shared__ __align__(32) unsigned short As[GBM * GBK];
  __shared__ __align__(32) unsigned short Bs[GBN * GBK];  // stored [n][k]

  const int tid  = threadIdx.x;
  const int wave = tid >> 5;
  const int lane = tid & 31;
  const int bm = blockIdx.y * GBM;
  const int bn = blockIdx.x * GBN;
  const int wm = (wave >> 1) * 32;
  const int wn = (wave & 1) * 32;
  const int r  = lane & 15;
  const int hi = lane >> 4;

  const int arow = tid >> 1;          // A: (row, 16-wide k chunk)
  const int akq  = (tid & 1) * 16;
  const int bnn  = tid & 63;          // B: (col n, 8-deep k chunk)
  const int bkq  = (tid >> 6) * 8;

  v8f acc[2][2] = {};

  for (int k0 = 0; k0 < K; k0 += GBK) {
    if constexpr (ABF) {
      const unsigned short* ap =
          (const unsigned short*)Av + (size_t)(bm + arow) * lda + k0 + akq;
      cpy16(ap, &As[arow * GBK + akq]);
      if (k0 + GBK < K) __builtin_prefetch(ap + GBK, 0, 0);
    } else {
      const float* ap = (const float*)Av + (size_t)(bm + arow) * lda + k0 + akq;
      cvt16(ap, &As[arow * GBK + akq]);
      if (k0 + GBK < K) __builtin_prefetch(ap + GBK, 0, 0);
    }

    if constexpr (BT) {
      // B^T [N][K] bf16: contiguous 8 u16 (16B) per thread, pure copy
      const unsigned short* bp = B + (size_t)(bn + bnn) * ldb + k0 + bkq;
      *(v4u*)&Bs[bnn * GBK + bkq] = *(const v4u*)bp;
    } else {
      // B [K][N] bf16: 8 strided u16 loads, pack pairs
      const unsigned short* bp = B + (size_t)(k0 + bkq) * ldb + bn + bnn;
      v4u u = {(unsigned)bp[0]                 | ((unsigned)bp[(size_t)ldb] << 16),
               (unsigned)bp[2 * (size_t)ldb]   | ((unsigned)bp[3 * (size_t)ldb] << 16),
               (unsigned)bp[4 * (size_t)ldb]   | ((unsigned)bp[5 * (size_t)ldb] << 16),
               (unsigned)bp[6 * (size_t)ldb]   | ((unsigned)bp[7 * (size_t)ldb] << 16)};
      *(v4u*)&Bs[bnn * GBK + bkq] = u;
    }
    __syncthreads();

    #pragma unroll
    for (int mi = 0; mi < 2; ++mi) {
      v16bf a = ldA(As, GBK, wm + mi * 16, r, hi, 0);
      #pragma unroll
      for (int ni = 0; ni < 2; ++ni) {
        v16bf b = ldB(Bs, GBK, wn + ni * 16, r, hi, 0);
        acc[mi][ni] = __builtin_amdgcn_wmma_f32_16x16x32_bf16(
            false, a, false, b, (short)0, acc[mi][ni], false, false);
      }
    }
    __syncthreads();
  }

  #pragma unroll
  for (int mi = 0; mi < 2; ++mi) {
    #pragma unroll
    for (int ni = 0; ni < 2; ++ni) {
      int col = bn + wn + ni * 16 + r;
      float bv = bias ? bias[col] : 0.0f;
      #pragma unroll
      for (int i = 0; i < 8; ++i) {
        int row = bm + wm + mi * 16 + hi * 8 + i;
        float val = acc[mi][ni][i] * alpha + bv;
        if constexpr (CBF)
          ((unsigned short*)Cv)[(size_t)row * ldc + col] = f2bf(val);
        else
          ((float*)Cv)[(size_t)row * ldc + col] = val;
      }
    }
  }
}

// ---------------------------------------------------------------------------
// Fused per-head score kernel. blockIdx.z = head. Three WMMA accumulations
// (S = qh@kh^T, mu = qmu@kmu^T, lv = qvar@kvar^T); all inputs bf16 in ws so
// staging is pure b128 copies. Epilogue:
//   attn = S*scale (NT store) ; attention_pre = (eps*exp(0.5*lv)+mu)*scale.
// ---------------------------------------------------------------------------
#define SK 64

__global__ __launch_bounds__(256) void score_fused_kernel(
    const unsigned short* __restrict__ q,    const unsigned short* __restrict__ k,
    const unsigned short* __restrict__ qmu,  const unsigned short* __restrict__ kmu,
    const unsigned short* __restrict__ qvar, const unsigned short* __restrict__ kvar,
    const float* __restrict__ eps,
    float* __restrict__ attn_out, float* __restrict__ p_out) {
  __shared__ __align__(32) unsigned short Aq[128 * SK];
  __shared__ __align__(32) unsigned short Am[128 * SK];
  __shared__ __align__(32) unsigned short Av[128 * SK];
  __shared__ __align__(32) unsigned short Bk[64 * SK];   // [n][k]
  __shared__ __align__(32) unsigned short Bm[64 * SK];
  __shared__ __align__(32) unsigned short Bv[64 * SK];

  const int tid  = threadIdx.x;
  const int wave = tid >> 5;
  const int lane = tid & 31;
  const int h  = blockIdx.z;
  const int bm = blockIdx.y * 128;
  const int bn = blockIdx.x * 64;
  const int wm = (wave >> 1) * 32;
  const int wn = (wave & 1) * 32;
  const int r  = lane & 15;
  const int hi = lane >> 4;
  const int coff = h * DH_;

  {  // A tiles: 128x64 bf16, thread -> (row, 32-wide k chunk); pure copies
    const int row = tid >> 1;
    const int kq  = (tid & 1) * 32;
    size_t g = (size_t)(bm + row) * D_ + coff + kq;
    int li = row * SK + kq;
    cpy16(q    + g, &Aq[li]);  cpy16(q    + g + 16, &Aq[li + 16]);
    cpy16(qmu  + g, &Am[li]);  cpy16(qmu  + g + 16, &Am[li + 16]);
    cpy16(qvar + g, &Av[li]);  cpy16(qvar + g + 16, &Av[li + 16]);
  }
  {  // B tiles: 64x64 bf16 stored [n][k], thread -> (n, 16-wide k chunk)
    const int nn = tid & 63;
    const int kq = (tid >> 6) * 16;
    size_t g = (size_t)(bn + nn) * D_ + coff + kq;
    int li = nn * SK + kq;
    cpy16(k    + g, &Bk[li]);
    cpy16(kmu  + g, &Bm[li]);
    cpy16(kvar + g, &Bv[li]);
  }
  __syncthreads();

  v8f accS[2][2] = {}, accM[2][2] = {}, accV[2][2] = {};

  #pragma unroll
  for (int ks = 0; ks < SK; ks += 32) {
    #pragma unroll
    for (int mi = 0; mi < 2; ++mi) {
      v16bf aq = ldA(Aq, SK, wm + mi * 16, r, hi, ks);
      v16bf am = ldA(Am, SK, wm + mi * 16, r, hi, ks);
      v16bf av = ldA(Av, SK, wm + mi * 16, r, hi, ks);
      #pragma unroll
      for (int ni = 0; ni < 2; ++ni) {
        v16bf bk  = ldB(Bk, SK, wn + ni * 16, r, hi, ks);
        v16bf bm2 = ldB(Bm, SK, wn + ni * 16, r, hi, ks);
        v16bf bv  = ldB(Bv, SK, wn + ni * 16, r, hi, ks);
        accS[mi][ni] = __builtin_amdgcn_wmma_f32_16x16x32_bf16(
            false, aq, false, bk, (short)0, accS[mi][ni], false, false);
        accM[mi][ni] = __builtin_amdgcn_wmma_f32_16x16x32_bf16(
            false, am, false, bm2, (short)0, accM[mi][ni], false, false);
        accV[mi][ni] = __builtin_amdgcn_wmma_f32_16x16x32_bf16(
            false, av, false, bv, (short)0, accV[mi][ni], false, false);
      }
    }
  }

  const size_t hbase = (size_t)h * L_ * L_;
  #pragma unroll
  for (int mi = 0; mi < 2; ++mi) {
    #pragma unroll
    for (int ni = 0; ni < 2; ++ni) {
      int col = bn + wn + ni * 16 + r;
      #pragma unroll
      for (int i = 0; i < 8; ++i) {
        int row = bm + wm + mi * 16 + hi * 8 + i;
        size_t idx = hbase + (size_t)row * L_ + col;
        // attn: write-once, never re-read -> NT store
        __builtin_nontemporal_store(accS[mi][ni][i] * SCALE_, &attn_out[idx]);
        // eps: read-once stream -> NT load
        float e = __builtin_nontemporal_load(&eps[idx]);
        float p = e * __expf(0.5f * accV[mi][ni][i]) + accM[mi][ni][i];
        p_out[idx] = p * SCALE_;  // re-read by softmax -> RT
      }
    }
  }
}

// ---------------------------------------------------------------------------
// In-place row softmax over [H*L, L]; also emits packed-bf16 copy for the
// ctx GEMM (removes its per-tile conversions and halves its read traffic).
// One 256-thread block per row; 8 contiguous elements per thread (b128 I/O).
// ---------------------------------------------------------------------------
__global__ __launch_bounds__(256) void softmax_kernel(
    float* __restrict__ p, unsigned short* __restrict__ pb) {
  __shared__ float red[256];
  const int tid = threadIdx.x;
  float* pr = p + (size_t)blockIdx.x * L_;
  unsigned short* pbr = pb + (size_t)blockIdx.x * L_;

  v4f a = *(const v4f*)(pr + tid * 8);
  v4f b = *(const v4f*)(pr + tid * 8 + 4);
  float mx = fmaxf(fmaxf(fmaxf(a[0], a[1]), fmaxf(a[2], a[3])),
                   fmaxf(fmaxf(b[0], b[1]), fmaxf(b[2], b[3])));
  red[tid] = mx;
  __syncthreads();
  for (int s = 128; s > 0; s >>= 1) {
    if (tid < s) red[tid] = fmaxf(red[tid], red[tid + s]);
    __syncthreads();
  }
  mx = red[0];
  __syncthreads();

  float v[8];
  float sum = 0.0f;
  #pragma unroll
  for (int j = 0; j < 4; ++j) { v[j] = __expf(a[j] - mx); sum += v[j]; }
  #pragma unroll
  for (int j = 0; j < 4; ++j) { v[4 + j] = __expf(b[j] - mx); sum += v[4 + j]; }
  red[tid] = sum;
  __syncthreads();
  for (int s = 128; s > 0; s >>= 1) {
    if (tid < s) red[tid] += red[tid + s];
    __syncthreads();
  }
  float inv = 1.0f / red[0];
  #pragma unroll
  for (int j = 0; j < 8; ++j) v[j] *= inv;

  v4f o0 = {v[0], v[1], v[2], v[3]};
  v4f o1 = {v[4], v[5], v[6], v[7]};
  *(v4f*)(pr + tid * 8)     = o0;
  *(v4f*)(pr + tid * 8 + 4) = o1;
  v4u u = {pk2(v[0], v[1]), pk2(v[2], v[3]), pk2(v[4], v[5]), pk2(v[6], v[7])};
  *(v4u*)(pbr + tid * 8) = u;
}

// ---------------------------------------------------------------------------
extern "C" void kernel_launch(void* const* d_in, const int* in_sizes, int n_in,
                              void* d_out, int out_size, void* d_ws, size_t ws_size,
                              hipStream_t stream) {
  (void)in_sizes; (void)n_in; (void)out_size; (void)ws_size;
  const float* aq   = (const float*)d_in[0];
  const float* ak   = (const float*)d_in[1];
  const float* av   = (const float*)d_in[2];
  const float* eps  = (const float*)d_in[3];
  const float* Wq   = (const float*)d_in[4];
  const float* bq   = (const float*)d_in[5];
  const float* Wk   = (const float*)d_in[6];
  const float* bk   = (const float*)d_in[7];
  const float* Wv   = (const float*)d_in[8];
  const float* bv   = (const float*)d_in[9];
  const float* Wmu  = (const float*)d_in[10];
  const float* bmu  = (const float*)d_in[11];
  const float* Wvar = (const float*)d_in[12];
  const float* bvar = (const float*)d_in[13];
  const float* Wc   = (const float*)d_in[14];
  const float* bc   = (const float*)d_in[15];

  typedef unsigned short us;
  const size_t LD = (size_t)L_ * D_;   // 1,048,576 elems
  const size_t DD = (size_t)D_ * D_;   // 262,144 elems
  const size_t HLL = (size_t)H_ * L_ * L_;

  us* w = (us*)d_ws;
  us* qb    = w + 0 * LD;
  us* kb    = w + 1 * LD;
  us* vb    = w + 2 * LD;
  us* qmub  = w + 3 * LD;
  us* kmub  = w + 4 * LD;
  us* qvarb = w + 5 * LD;
  us* kvarb = w + 6 * LD;
  us* ctxb  = w + 7 * LD;
  us* WqT   = w + 8 * LD;
  us* WkT   = WqT   + DD;
  us* WvT   = WkT   + DD;
  us* WmuT  = WvT   + DD;
  us* WvarT = WmuT  + DD;
  us* WcT   = WvarT + DD;
  us* attb  = WcT   + DD;              // [H,L,L] bf16, 67MB

  float* out        = (float*)d_out;
  float* attn       = out + LD;        // [H,L,L]
  float* attention  = attn + HLL;      // [H,L,L]

  dim3 blk(256);
  dim3 gT(D_ / 32, D_ / 32);           // (16,16) weight transpose
  dim3 gProj(D_ / GBN, L_ / GBM);      // (8,16)

  // one-time weight transpose+convert to bf16 [N][K]
  transpose_cvt_kernel<<<gT, blk, 0, stream>>>(Wq,   WqT);
  transpose_cvt_kernel<<<gT, blk, 0, stream>>>(Wk,   WkT);
  transpose_cvt_kernel<<<gT, blk, 0, stream>>>(Wv,   WvT);
  transpose_cvt_kernel<<<gT, blk, 0, stream>>>(Wmu,  WmuT);
  transpose_cvt_kernel<<<gT, blk, 0, stream>>>(Wvar, WvarT);
  transpose_cvt_kernel<<<gT, blk, 0, stream>>>(Wc,   WcT);

  // input projections: f32 A, bf16 B^T, bf16 C
  gemm_bf16_kernel<false, true, true><<<gProj, blk, 0, stream>>>(
      aq, D_, WqT, D_, qb, D_, bq, 1.0f, L_, D_, D_);
  gemm_bf16_kernel<false, true, true><<<gProj, blk, 0, stream>>>(
      ak, D_, WkT, D_, kb, D_, bk, 1.0f, L_, D_, D_);
  gemm_bf16_kernel<false, true, true><<<gProj, blk, 0, stream>>>(
      av, D_, WvT, D_, vb, D_, bv, 1.0f, L_, D_, D_);

  // mu / var projections: bf16 A, bf16 B^T, bf16 C
  gemm_bf16_kernel<true, true, true><<<gProj, blk, 0, stream>>>(
      qb, D_, WmuT,  D_, qmub,  D_, bmu,  1.0f, L_, D_, D_);
  gemm_bf16_kernel<true, true, true><<<gProj, blk, 0, stream>>>(
      kb, D_, WmuT,  D_, kmub,  D_, bmu,  1.0f, L_, D_, D_);
  gemm_bf16_kernel<true, true, true><<<gProj, blk, 0, stream>>>(
      qb, D_, WvarT, D_, qvarb, D_, bvar, 1.0f, L_, D_, D_);
  gemm_bf16_kernel<true, true, true><<<gProj, blk, 0, stream>>>(
      kb, D_, WvarT, D_, kvarb, D_, bvar, 1.0f, L_, D_, D_);

  // fused attn / mu / logvar / reparameterize
  dim3 gScore(L_ / 64, L_ / 128, H_);  // (32,16,8)
  score_fused_kernel<<<gScore, blk, 0, stream>>>(
      qb, kb, qmub, kmub, qvarb, kvarb, eps, attn, attention);

  // in-place softmax (+ bf16 side copy for ctx GEMM)
  softmax_kernel<<<dim3(H_ * L_), blk, 0, stream>>>(attention, attb);

  // ctx_h = attention_h @ v_h : bf16 A (attb), bf16 B [K][N] (vb), bf16 C
  for (int h = 0; h < H_; ++h) {
    gemm_bf16_kernel<true, false, true><<<dim3(1, L_ / GBM), blk, 0, stream>>>(
        attb + (size_t)h * L_ * L_, L_,
        vb + h * DH_, D_,
        ctxb + h * DH_, D_,
        nullptr, 1.0f, L_, DH_, L_);
  }

  // output projection: bf16 A, bf16 B^T, f32 C + bias
  gemm_bf16_kernel<true, true, false><<<gProj, blk, 0, stream>>>(
      ctxb, D_, WcT, D_, out, D_, bc, 1.0f, L_, D_, D_);
}
